// SelfAttention_21002390077562
// MI455X (gfx1250) — matrix operand order repaired and checked by reading, not compiled, not run on hardware
//
#include <hip/hip_runtime.h>
#include <hip/hip_bf16.h>

typedef __attribute__((ext_vector_type(16))) _Float16 v16h;
typedef __attribute__((ext_vector_type(8)))  _Float16 h8;
typedef __attribute__((ext_vector_type(8)))  float    v8f;

#define B_  4
#define C_  256
#define CQ_ 32
#define N_  4096

static __device__ inline v16h make16(h8 lo, h8 hi) {
  v16h r;
#pragma unroll
  for (int i = 0; i < 8; ++i) { r[i] = lo[i]; r[8 + i] = hi[i]; }
  return r;
}

static __device__ inline v8f wmma_f16(v16h a, v16h b, v8f c) {
  // D = A(16x32 f16) * B(32x16 f16) + C(16x16 f32)
  return __builtin_amdgcn_wmma_f32_16x16x32_f16(false, a, false, b, (short)0, c,
                                                false, false);
}

// ---------------------------------------------------------------------------
// Kernel 1a: Q/K projections. qT/kT stored as [b][n][32ch] f16 (ch contiguous)
// so the attention kernel's A- and B-tile fetches are contiguous b128 loads.
// ---------------------------------------------------------------------------
__global__ __launch_bounds__(256) void proj_qk_kernel(
    const float* __restrict__ x, const float* __restrict__ wq,
    const float* __restrict__ bq, const float* __restrict__ wk,
    const float* __restrict__ bk, _Float16* __restrict__ qT,
    _Float16* __restrict__ kT) {
  int t = blockIdx.x * 256 + threadIdx.x;
  int b = t >> 12, n = t & (N_ - 1);
  const float* xp = x + (size_t)b * C_ * N_ + n;
  float qa[CQ_], ka[CQ_];
#pragma unroll
  for (int o = 0; o < CQ_; ++o) { qa[o] = bq[o]; ka[o] = bk[o]; }
  for (int c = 0; c < C_; ++c) {
    float xv = xp[(size_t)c * N_];
#pragma unroll
    for (int o = 0; o < CQ_; ++o) {
      qa[o] = fmaf(wq[o * C_ + c], xv, qa[o]);
      ka[o] = fmaf(wk[o * C_ + c], xv, ka[o]);
    }
  }
  _Float16* qp = qT + ((size_t)b * N_ + n) * CQ_;
  _Float16* kp = kT + ((size_t)b * N_ + n) * CQ_;
#pragma unroll
  for (int s = 0; s < 4; ++s) {
    h8 hq, hk;
#pragma unroll
    for (int i = 0; i < 8; ++i) {
      hq[i] = (_Float16)qa[s * 8 + i];
      hk[i] = (_Float16)ka[s * 8 + i];
    }
    *(h8*)(qp + s * 8) = hq;
    *(h8*)(kp + s * 8) = hk;
  }
}

// ---------------------------------------------------------------------------
// Kernel 1b: V projection. vf stored as [b][c][n] f16 (n contiguous) -> the
// attention kernel's V A-tiles (M=c, K=j) are contiguous per-lane loads.
// ---------------------------------------------------------------------------
__global__ __launch_bounds__(256) void proj_v_kernel(
    const float* __restrict__ x, const float* __restrict__ wv,
    const float* __restrict__ bv, _Float16* __restrict__ vf) {
  int b = blockIdx.x >> 4, nb = blockIdx.x & 15;
  int n = nb * 256 + threadIdx.x;
  const float* xp = x + (size_t)b * C_ * N_ + n;
  for (int co = 0; co < C_; co += 16) {
    float acc[16];
#pragma unroll
    for (int u = 0; u < 16; ++u) acc[u] = bv[co + u];
    for (int c = 0; c < C_; ++c) {
      float xv = xp[(size_t)c * N_];
#pragma unroll
      for (int u = 0; u < 16; ++u)
        acc[u] = fmaf(wv[(co + u) * C_ + c], xv, acc[u]);
    }
#pragma unroll
    for (int u = 0; u < 16; ++u)
      vf[((size_t)b * C_ + co + u) * N_ + n] = (_Float16)acc[u];
  }
}

// ---------------------------------------------------------------------------
// Kernel 2: fused energy -> softmax -> attention write -> out = V*A^T + resid.
// Block = (batch, 64 query rows). 8 waves = 4 i-tiles x 2 c-halves.
// Three j-sweeps: rowmax, rowsum(exp), normalize+write+AV-GEMM.
// No inter-wave barriers; per-wave 1KB LDS slice for the att D->B transpose.
// ---------------------------------------------------------------------------
__global__ __launch_bounds__(256) void attn_kernel(
    const _Float16* __restrict__ qT, const _Float16* __restrict__ kT,
    const _Float16* __restrict__ vf, const float* __restrict__ x,
    const float* __restrict__ gamma, float* __restrict__ outp,
    float* __restrict__ attp) {
  __shared__ _Float16 ldsA[8][16 * 32];
  const int b     = blockIdx.x >> 6;
  const int iblk  = blockIdx.x & 63;
  const int w     = threadIdx.x >> 5;
  const int lane  = threadIdx.x & 31;
  const int itile = w >> 1, chalf = w & 1;
  const int i0  = iblk * 64 + itile * 16;
  const int c0  = chalf * 128;
  const int lhi = lane >> 4, llo = lane & 15;
  _Float16* lds = &ldsA[w][0];
  const v8f vzero = {};

  // Q A-tile (A 16x32 f16): lane l -> M=l%16; K=ch 16*hi..16*hi+15
  const _Float16* qp = qT + ((size_t)b * N_ + i0 + llo) * CQ_ + 16 * lhi;
  const v16h aq = make16(*(const h8*)qp, *(const h8*)(qp + 8));

  // K B-tile base (B 32x16 f16): lane l -> N=j0+l%16; K=ch 16*hi..16*hi+15
  const _Float16* kbase = kT + ((size_t)b * N_ + llo) * CQ_ + 16 * lhi;

  // ---- sweep 1: row max ----
  float m[8];
#pragma unroll
  for (int r = 0; r < 8; ++r) m[r] = -3.0e38f;
  for (int jt = 0; jt < N_ / 16; ++jt) {
    const _Float16* kp = kbase + (size_t)jt * 16 * CQ_;
    v16h bk = make16(*(const h8*)kp, *(const h8*)(kp + 8));
    v8f e = wmma_f16(aq, bk, vzero);
#pragma unroll
    for (int r = 0; r < 8; ++r) m[r] = fmaxf(m[r], e[r]);
  }
#pragma unroll
  for (int r = 0; r < 8; ++r) {
#pragma unroll
    for (int msk = 1; msk < 16; msk <<= 1)
      m[r] = fmaxf(m[r], __shfl_xor(m[r], msk, 32));
  }

  // ---- sweep 2: row sum of exp ----
  float s[8];
#pragma unroll
  for (int r = 0; r < 8; ++r) s[r] = 0.0f;
  for (int jt = 0; jt < N_ / 16; ++jt) {
    const _Float16* kp = kbase + (size_t)jt * 16 * CQ_;
    v16h bk = make16(*(const h8*)kp, *(const h8*)(kp + 8));
    v8f e = wmma_f16(aq, bk, vzero);
#pragma unroll
    for (int r = 0; r < 8; ++r) s[r] += __expf(e[r] - m[r]);
  }
  float inv[8];
#pragma unroll
  for (int r = 0; r < 8; ++r) {
#pragma unroll
    for (int msk = 1; msk < 16; msk <<= 1) s[r] += __shfl_xor(s[r], msk, 32);
    inv[r] = 1.0f / s[r];
  }

  // ---- sweep 3: normalize, write attention once, fused AV GEMM ----
  v8f acc[8];
#pragma unroll
  for (int ct = 0; ct < 8; ++ct) acc[ct] = vzero;
  // V A-tile (A 16x32 f16): M=c=llo within tile; K runs j0+{8*hi, 16+8*hi}
  const _Float16* vbase = vf + ((size_t)b * C_ + c0 + llo) * N_ + 8 * lhi;
  float* attb = attp + ((size_t)b << 24);
  const float gam = gamma[0];

  for (int jc = 0; jc < N_ / 32; ++jc) {
    const int j0 = jc * 32;
    const _Float16* kp0 = kbase + (size_t)j0 * CQ_;
    const _Float16* kp1 = kbase + (size_t)(j0 + 16) * CQ_;
    v16h bk0 = make16(*(const h8*)kp0, *(const h8*)(kp0 + 8));
    v16h bk1 = make16(*(const h8*)kp1, *(const h8*)(kp1 + 8));
    v8f e0 = wmma_f16(aq, bk0, vzero);
    v8f e1 = wmma_f16(aq, bk1, vzero);
    float a0[8], a1[8];
#pragma unroll
    for (int r = 0; r < 8; ++r) {
      a0[r] = __expf(e0[r] - m[r]) * inv[r];
      a1[r] = __expf(e1[r] - m[r]) * inv[r];
    }
    if (chalf == 0) {  // wave-uniform: only one c-half writes attention
#pragma unroll
      for (int r = 0; r < 8; ++r) {
        const size_t row = (size_t)(i0 + r + 8 * lhi) << 12;
        attb[row + j0 + llo]      = a0[r];
        attb[row + j0 + 16 + llo] = a1[r];
      }
    }
    // D-layout (i=r+8*hi, j=16t+llo) -> LDS [i][j] f16
#pragma unroll
    for (int r = 0; r < 8; ++r) {
      const int il = r + 8 * lhi;
      lds[il * 32 + llo]      = (_Float16)a0[r];
      lds[il * 32 + 16 + llo] = (_Float16)a1[r];
    }
    // att B-tile (B 32x16): lane l -> N=i=llo; K=j runs {8*hi.., 16+8*hi..}
    const _Float16* lp = lds + llo * 32 + 8 * lhi;
    v16h batt = make16(*(const h8*)lp, *(const h8*)(lp + 16));
#pragma unroll
    for (int ct = 0; ct < 8; ++ct) {
      const _Float16* vp = vbase + (size_t)(ct * 16) * N_ + j0;
      v16h av = make16(*(const h8*)vp, *(const h8*)(vp + 16));
      acc[ct] = wmma_f16(av, batt, acc[ct]);  // D[c,i] += V[c,j]*att[i,j]
    }
  }

  // ---- epilogue: out = gamma*attn_out + x (coalesced in i) ----
  const float* xb = x + ((size_t)b * C_ + c0) * N_ + i0;
  float*       ob = outp + ((size_t)b * C_ + c0) * N_ + i0;
#pragma unroll
  for (int ct = 0; ct < 8; ++ct) {
#pragma unroll
    for (int r = 0; r < 8; ++r) {
      const size_t off = (size_t)(ct * 16 + r + 8 * lhi) * N_ + llo;
      ob[off] = fmaf(gam, acc[ct][r], xb[off]);
    }
  }
}

// ---------------------------------------------------------------------------
extern "C" void kernel_launch(void* const* d_in, const int* in_sizes, int n_in,
                              void* d_out, int out_size, void* d_ws,
                              size_t ws_size, hipStream_t stream) {
  const float* x     = (const float*)d_in[0];
  const float* wq    = (const float*)d_in[1];
  const float* bq    = (const float*)d_in[2];
  const float* wk    = (const float*)d_in[3];
  const float* bk    = (const float*)d_in[4];
  const float* wv    = (const float*)d_in[5];
  const float* bv    = (const float*)d_in[6];
  const float* gamma = (const float*)d_in[7];

  // workspace: qT (1MB) | kT (1MB) | vf (8MB), all f16
  _Float16* qT = (_Float16*)d_ws;
  _Float16* kT = qT + (size_t)B_ * N_ * CQ_;
  _Float16* vf = kT + (size_t)B_ * N_ * CQ_;

  float* outp = (float*)d_out;                       // [B,C,W,H]
  float* attp = outp + (size_t)B_ * C_ * N_;         // [B,N,N]

  proj_qk_kernel<<<(B_ * N_) / 256, 256, 0, stream>>>(x, wq, bq, wk, bk, qT, kT);
  proj_v_kernel<<<B_ * (N_ / 256), 256, 0, stream>>>(x, wv, bv, vf);
  attn_kernel<<<B_ * (N_ / 64), 256, 0, stream>>>(qT, kT, vf, x, gamma, outp,
                                                  attp);
}
